// HypergraphTransformerBlock_21861383537286
// MI455X (gfx1250) — matrix operand order repaired and checked by reading, not compiled, not run on hardware
//
#include <hip/hip_runtime.h>
#include <hip/hip_bf16.h>
#include <math.h>

// ---------------------------------------------------------------------------
// HypergraphConv + TransformerEncoderLayer block for MI455X (gfx1250)
// - bf16 WMMA (v_wmma_f32_16x16x32_bf16) for all dense math
// - weights pre-converted to bf16 in global memory (halves weight HBM traffic)
// - GEMM W-tiles staged with GLOBAL_LOAD_ASYNC_TO_LDS_B128 (ASYNCcnt path)
// - flash attention (online softmax, scores never hit HBM)
// - atomic segment ops for the sparse hypergraph aggregation
// ---------------------------------------------------------------------------

typedef __bf16 bf16_t;
typedef __attribute__((ext_vector_type(16))) __bf16 bf16x16;
typedef __attribute__((ext_vector_type(8)))  float  f32x8;

#define N_NODES 4096
#define M_EDGES 1024
#define NNZ_E   65536
#define DMODEL  512
#define NHEAD   8
#define DHEAD   64
#define DFF_    2048

// float -> bf16 with round-to-nearest-even (bit manipulation; no __bf16 arith)
__device__ __forceinline__ bf16_t f2bf(float f) {
  union { float f; unsigned u; } in; in.f = f;
  unsigned r = (in.u + 0x7FFFu + ((in.u >> 16) & 1u)) >> 16;
  union { unsigned short s; bf16_t b; } out; out.s = (unsigned short)r;
  return out.b;
}

// A-matrix (16x32 bf16) element->K mapping per CDNA5 ISA 7.12.2:
// V0:K0,1 V1:K2,3 V2:K4,5 V3:K6,7 (half0) / +8 (half1); V4..7: +16.
__device__ __forceinline__ int a_k(int i, int half) {
  int v = i >> 1, p = i & 1;
  int k = (v < 4) ? (2 * v + p) : (16 + 2 * (v - 4) + p);
  return k + 8 * half;
}
// B-matrix (32x16 bf16): same element->K mapping as the A-matrix.
__device__ __forceinline__ int b_k(int i, int half) { return a_k(i, half); }

// atomic float max supporting negative values
__device__ __forceinline__ void atomicMaxFloat(float* addr, float val) {
  if (val >= 0.0f) atomicMax((int*)addr, __float_as_int(val));
  else             atomicMin((unsigned int*)addr, __float_as_uint(val));
}

// ---------------------------------------------------------------------------
// WMMA GEMM: C[M,N] = A[M,K] @ W[N,K]^T (+bias, optional ReLU)
// A: f32 activations (converted to bf16 into LDS), W: bf16 weights.
// block = 256 threads = 8 waves (4 M-rows x 2 N-cols of waves).
// Block tile 64(M) x 128(N); each wave owns 16x64 (4 accumulators).
// W tile (8KB/chunk) staged via async global->LDS b128 copies.
// ---------------------------------------------------------------------------
// single merged LDS block so byte offsets are known for the async path:
//   [0 .. 4095]     lA : 64x32 bf16  (4KB)
//   [4096 .. 12287] lW : 128x32 bf16 (8KB)
#define LW_BYTE_BASE 4096u

template <bool RELU>
__global__ __launch_bounds__(256)
void gemm_bf16_kernel(const float* __restrict__ A, const bf16_t* __restrict__ W,
                      const float* __restrict__ bias, float* __restrict__ C,
                      int M, int N, int K) {
  __shared__ bf16_t smem[64 * 32 + 128 * 32];
  bf16_t* lA = smem;
  bf16_t* lW = smem + 64 * 32;
  const int tid  = threadIdx.x;
  const int lane = tid & 31;
  const int wave = tid >> 5;
  const int wm = wave >> 1;          // 0..3 : wave row (16 M each)
  const int wn = wave & 1;           // 0..1 : wave col (64 N each)
  const int m0 = blockIdx.x * 64;
  const int n0 = blockIdx.y * 128;
  const int row16 = lane & 15, half = lane >> 4;

  f32x8 acc[4];
#pragma unroll
  for (int t = 0; t < 4; t++)
#pragma unroll
    for (int r = 0; r < 8; r++) acc[t][r] = 0.0f;

  for (int k0 = 0; k0 < K; k0 += 32) {
    // stage A chunk: 64x32 f32 -> bf16, 8 elems/thread
    for (int t = tid; t < 64 * 32; t += 256) {
      int r = t >> 5, c = t & 31;
      lA[t] = f2bf(A[(size_t)(m0 + r) * K + k0 + c]);
    }
    // stage W chunk: 128x32 bf16 = 512 x 16B segments; 2 async b128 per thread
#if defined(__gfx1250__)
#pragma unroll
    for (int j = 0; j < 2; j++) {
      int t = tid + j * 256;                 // segment id
      int wr = t >> 2, seg = t & 3;          // row, 16B segment in row
      unsigned lds_off = LW_BYTE_BASE + (unsigned)t * 16u;
      unsigned long long ga =
          (unsigned long long)(size_t)(W + (size_t)(n0 + wr) * K + k0 + seg * 8);
      asm volatile("global_load_async_to_lds_b128 %0, %1, off"
                   :: "v"(lds_off), "v"(ga) : "memory");
    }
#else
    for (int t = tid; t < 128 * 32; t += 256) {
      int r = t >> 5, c = t & 31;
      lW[t] = W[(size_t)(n0 + r) * K + k0 + c];
    }
#endif
    // prefetch next A chunk: one 128B cacheline (= one k-row) per thread
    if (k0 + 32 < K && tid < 64)
      __builtin_prefetch(&A[(size_t)(m0 + tid) * K + k0 + 32], 0, 3);
#if defined(__gfx1250__)
    asm volatile("s_wait_asynccnt 0" ::: "memory");
#endif
    __syncthreads();

    bf16x16 af;
#pragma unroll
    for (int i = 0; i < 16; i++)
      af[i] = lA[(wm * 16 + row16) * 32 + a_k(i, half)];
#pragma unroll
    for (int t = 0; t < 4; t++) {
      bf16x16 bfr;
#pragma unroll
      for (int i = 0; i < 16; i++)
        bfr[i] = lW[(wn * 64 + t * 16 + row16) * 32 + b_k(i, half)];
      acc[t] = __builtin_amdgcn_wmma_f32_16x16x32_bf16(false, af, false, bfr,
                                                       (short)0, acc[t], false, false);
    }
    __syncthreads();
  }

#pragma unroll
  for (int t = 0; t < 4; t++) {
    const int gcol = n0 + wn * 64 + t * 16 + row16;
    const float bv = bias ? bias[gcol] : 0.0f;
#pragma unroll
    for (int r = 0; r < 8; r++) {
      int grow = m0 + wm * 16 + r + 8 * half;
      float v = acc[t][r] + bv;
      if (RELU) v = fmaxf(v, 0.0f);
      C[(size_t)grow * N + gcol] = v;
    }
  }
}

// ---------------------------------------------------------------------------
// Flash attention: grid = (N/128, H); 8 waves, each owns a 16-row Q tile.
// ---------------------------------------------------------------------------
__global__ __launch_bounds__(256)
void attn_kernel(const float* __restrict__ qkv, float* __restrict__ ctx) {
  const int h   = blockIdx.y;
  const int qb  = blockIdx.x * 128;
  const int tid = threadIdx.x, lane = tid & 31, wave = tid >> 5;
  const int row16 = lane & 15, half = lane >> 4;
  const int LD = 3 * DMODEL;

  __shared__ bf16_t lK[32 * 64];
  __shared__ bf16_t lV[32 * 64];
  __shared__ float  sS[8][16 * 32];
  __shared__ bf16_t sP[8][16 * 32];
  __shared__ float  sM[8][16], sL[8][16], sC[8][16];

  // Q fragments: 16 rows x 64 head-dims -> two 16x32 bf16 A-fragments
  bf16x16 qf[2];
  const int qrow = qb + wave * 16 + row16;
#pragma unroll
  for (int ch = 0; ch < 2; ch++)
#pragma unroll
    for (int i = 0; i < 16; i++)
      qf[ch][i] = f2bf(qkv[(size_t)qrow * LD + h * DHEAD + ch * 32 + a_k(i, half)]);

  f32x8 o[4];
#pragma unroll
  for (int t = 0; t < 4; t++)
#pragma unroll
    for (int r = 0; r < 8; r++) o[t][r] = 0.0f;

  if (lane < 16) { sM[wave][lane] = -1e30f; sL[wave][lane] = 0.0f; }

  for (int kb = 0; kb < N_NODES; kb += 32) {
    for (int t = tid; t < 32 * 64; t += 256) {
      int kr = t >> 6, kc = t & 63;
      lK[t] = f2bf(qkv[(size_t)(kb + kr) * LD + DMODEL     + h * DHEAD + kc]);
      lV[t] = f2bf(qkv[(size_t)(kb + kr) * LD + 2 * DMODEL + h * DHEAD + kc]);
    }
    __syncthreads();

    // S = (Q @ K^T) / 8 : two 16x16 tiles, each 2 WMMAs over head-dim
#pragma unroll
    for (int t = 0; t < 2; t++) {
      f32x8 s = {};
#pragma unroll
      for (int ch = 0; ch < 2; ch++) {
        bf16x16 bfr;
#pragma unroll
        for (int i = 0; i < 16; i++)
          bfr[i] = lK[(t * 16 + row16) * 64 + ch * 32 + b_k(i, half)];
        s = __builtin_amdgcn_wmma_f32_16x16x32_bf16(false, qf[ch], false, bfr,
                                                    (short)0, s, false, false);
      }
#pragma unroll
      for (int r = 0; r < 8; r++)
        sS[wave][(r + 8 * half) * 32 + t * 16 + row16] = s[r] * 0.125f;
    }
    __syncthreads();

    // online softmax (lanes 0..15 of each wave handle one row each)
    if (lane < 16) {
      int r = lane;
      float m = sM[wave][r], cmax = -1e30f;
      for (int c = 0; c < 32; c++) cmax = fmaxf(cmax, sS[wave][r * 32 + c]);
      float nm = fmaxf(m, cmax);
      float corr = __expf(m - nm), lsum = 0.0f;
      for (int c = 0; c < 32; c++) {
        float p = __expf(sS[wave][r * 32 + c] - nm);
        lsum += p;
        sP[wave][r * 32 + c] = f2bf(p);
      }
      sM[wave][r] = nm;
      sL[wave][r] = sL[wave][r] * corr + lsum;
      sC[wave][r] = corr;
    }
    __syncthreads();

    // O = O*corr + P @ V
    bf16x16 pf;
#pragma unroll
    for (int i = 0; i < 16; i++)
      pf[i] = sP[wave][row16 * 32 + a_k(i, half)];
#pragma unroll
    for (int t = 0; t < 4; t++) {
#pragma unroll
      for (int r = 0; r < 8; r++) o[t][r] *= sC[wave][r + 8 * half];
      bf16x16 vfr;
#pragma unroll
      for (int i = 0; i < 16; i++)
        vfr[i] = lV[b_k(i, half) * 64 + t * 16 + row16];
      o[t] = __builtin_amdgcn_wmma_f32_16x16x32_bf16(false, pf, false, vfr,
                                                     (short)0, o[t], false, false);
    }
    __syncthreads();
  }

#pragma unroll
  for (int t = 0; t < 4; t++)
#pragma unroll
    for (int r = 0; r < 8; r++) {
      int grow = qb + wave * 16 + r + 8 * half;
      float denom = sL[wave][r + 8 * half];
      ctx[(size_t)grow * DMODEL + h * DHEAD + t * 16 + row16] = o[t][r] / denom;
    }
}

// ---------------------------------------------------------------------------
// out = LayerNorm(a + b? + bias?) * g + be   — one block per row, D=512
// ---------------------------------------------------------------------------
__global__ __launch_bounds__(256)
void add_ln_kernel(const float* __restrict__ a, const float* __restrict__ b,
                   const float* __restrict__ bias, const float* __restrict__ g,
                   const float* __restrict__ be, float* __restrict__ out) {
  const int row = blockIdx.x, tid = threadIdx.x;
  __shared__ float red[8];
  float v[2];
#pragma unroll
  for (int j = 0; j < 2; j++) {
    int d = tid + j * 256;
    float x = a[(size_t)row * DMODEL + d];
    if (b)    x += b[(size_t)row * DMODEL + d];
    if (bias) x += bias[d];
    v[j] = x;
  }
  float s = v[0] + v[1];
  for (int off = 16; off > 0; off >>= 1) s += __shfl_xor(s, off, 32);
  if ((tid & 31) == 0) red[tid >> 5] = s;
  __syncthreads();
  float mean = 0.0f;
#pragma unroll
  for (int i = 0; i < 8; i++) mean += red[i];
  mean *= (1.0f / DMODEL);
  __syncthreads();
  float d0 = v[0] - mean, d1 = v[1] - mean;
  float q = d0 * d0 + d1 * d1;
  for (int off = 16; off > 0; off >>= 1) q += __shfl_xor(q, off, 32);
  if ((tid & 31) == 0) red[tid >> 5] = q;
  __syncthreads();
  float var = 0.0f;
#pragma unroll
  for (int i = 0; i < 8; i++) var += red[i];
  var *= (1.0f / DMODEL);
  float inv = rsqrtf(var + 1e-5f);
#pragma unroll
  for (int j = 0; j < 2; j++) {
    int d = tid + j * 256;
    out[(size_t)row * DMODEL + d] = (v[j] - mean) * inv * g[d] + be[d];
  }
}

// ---------------------------------------------------------------------------
// Elementwise f32 -> bf16 (weight pre-conversion, once per launch)
// ---------------------------------------------------------------------------
__global__ __launch_bounds__(256)
void cvt_bf16_kernel(const float* __restrict__ src, bf16_t* __restrict__ dst,
                     int n) {
  int i = blockIdx.x * 256 + threadIdx.x;
  if (i < n) dst[i] = f2bf(src[i]);
}

// ---------------------------------------------------------------------------
// Sparse hypergraph helpers
// ---------------------------------------------------------------------------
__global__ __launch_bounds__(256)
void fill_kernel(float* __restrict__ p, float val, int n) {
  int i = blockIdx.x * 256 + threadIdx.x;
  if (i < n) p[i] = val;
}

// out[row] = dot(X[row,:], att[0:512]); one wave per row
__global__ __launch_bounds__(256)
void rowdot_kernel(const float* __restrict__ X, const float* __restrict__ att,
                   float* __restrict__ out, int nrows) {
  int row = blockIdx.x * 8 + (threadIdx.x >> 5);
  int lane = threadIdx.x & 31;
  if (row >= nrows) return;
  float s = 0.0f;
  for (int d = lane; d < DMODEL; d += 32) s += X[(size_t)row * DMODEL + d] * att[d];
  for (int off = 16; off > 0; off >>= 1) s += __shfl_xor(s, off, 32);
  if (lane == 0) out[row] = s;
}

__global__ __launch_bounds__(256)
void edge_score_kernel(const int* __restrict__ node, const int* __restrict__ edge,
                       const float* __restrict__ sx, const float* __restrict__ se,
                       float* __restrict__ a, float* __restrict__ Dn,
                       float* __restrict__ Bn, float* __restrict__ amax) {
  int i = blockIdx.x * 256 + threadIdx.x;
  if (i >= NNZ_E) return;
  int n = node[i], e = edge[i];
  float v = sx[n] + se[e];
  v = (v >= 0.0f) ? v : 0.2f * v;  // leaky_relu(0.2)
  a[i] = v;
  atomicMaxFloat(&amax[n], v);
  atomicAdd(&Dn[n], 1.0f);
  atomicAdd(&Bn[e], 1.0f);
}

__global__ __launch_bounds__(256)
void edge_exp_kernel(const int* __restrict__ node, const float* __restrict__ a,
                     const float* __restrict__ amax, float* __restrict__ aexp,
                     float* __restrict__ den) {
  int i = blockIdx.x * 256 + threadIdx.x;
  if (i >= NNZ_E) return;
  float e = __expf(a[i] - amax[node[i]]);
  aexp[i] = e;
  atomicAdd(&den[node[i]], e);
}

// ef[edge] += Binv*alpha * xl[node]   (one block per incidence)
__global__ __launch_bounds__(256)
void scatter_edge_kernel(const int* __restrict__ node, const int* __restrict__ edge,
                         const float* __restrict__ aexp, const float* __restrict__ den,
                         const float* __restrict__ Bn, const float* __restrict__ xl,
                         float* __restrict__ ef) {
  int i = blockIdx.x;
  int n = node[i], e = edge[i];
  float alpha = aexp[i] / (den[n] + 1e-16f);
  float binv  = (Bn[e] > 0.0f) ? 1.0f / Bn[e] : 0.0f;
  float cf = binv * alpha;
  for (int d = threadIdx.x; d < DMODEL; d += 256)
    atomicAdd(&ef[(size_t)e * DMODEL + d], cf * xl[(size_t)n * DMODEL + d]);
}

// conv[node] += Dinv*alpha * ef[edge]
__global__ __launch_bounds__(256)
void scatter_node_kernel(const int* __restrict__ node, const int* __restrict__ edge,
                         const float* __restrict__ aexp, const float* __restrict__ den,
                         const float* __restrict__ Dn, const float* __restrict__ ef,
                         float* __restrict__ conv) {
  int i = blockIdx.x;
  int n = node[i], e = edge[i];
  float alpha = aexp[i] / (den[n] + 1e-16f);
  float dinv  = (Dn[n] > 0.0f) ? 1.0f / Dn[n] : 0.0f;
  float cf = dinv * alpha;
  for (int d = threadIdx.x; d < DMODEL; d += 256)
    atomicAdd(&conv[(size_t)n * DMODEL + d], cf * ef[(size_t)e * DMODEL + d]);
}

// ---------------------------------------------------------------------------
extern "C" void kernel_launch(void* const* d_in, const int* in_sizes, int n_in,
                              void* d_out, int out_size, void* d_ws, size_t ws_size,
                              hipStream_t stream) {
  const float* x      = (const float*)d_in[0];
  const int*   hidx   = (const int*)d_in[1];
  const float* hattr  = (const float*)d_in[2];
  const float* Wh     = (const float*)d_in[3];
  const float* att    = (const float*)d_in[4];
  const float* bias_h = (const float*)d_in[5];
  const float* ln1_g  = (const float*)d_in[6];
  const float* ln1_b  = (const float*)d_in[7];
  const float* ln2_g  = (const float*)d_in[8];
  const float* ln2_b  = (const float*)d_in[9];
  const float* in_w   = (const float*)d_in[10];
  const float* in_b   = (const float*)d_in[11];
  const float* out_w  = (const float*)d_in[12];
  const float* out_b  = (const float*)d_in[13];
  const float* ff1_w  = (const float*)d_in[14];
  const float* ff1_b  = (const float*)d_in[15];
  const float* ff2_w  = (const float*)d_in[16];
  const float* ff2_b  = (const float*)d_in[17];
  const float* tln1_g = (const float*)d_in[18];
  const float* tln1_b = (const float*)d_in[19];
  const float* tln2_g = (const float*)d_in[20];
  const float* tln2_b = (const float*)d_in[21];
  const int* node = hidx;
  const int* edge = hidx + NNZ_E;
  float* out = (float*)d_out;

  // workspace bump allocator (256B aligned)
  char* wptr = (char*)d_ws;
  auto alloc = [&](size_t bytes) -> void* {
    void* p = (void*)wptr;
    wptr += (bytes + 255) & ~(size_t)255;
    return p;
  };
  float* xl    = (float*)alloc((size_t)N_NODES * DMODEL * 4);
  float* el    = (float*)alloc((size_t)M_EDGES * DMODEL * 4);
  float* sx    = (float*)alloc((size_t)N_NODES * 4);
  float* se    = (float*)alloc((size_t)M_EDGES * 4);
  float* ascr  = (float*)alloc((size_t)NNZ_E * 4);
  float* aexp  = (float*)alloc((size_t)NNZ_E * 4);
  float* amax  = (float*)alloc((size_t)N_NODES * 4);
  float* den   = (float*)alloc((size_t)N_NODES * 4);
  float* Dn    = (float*)alloc((size_t)N_NODES * 4);
  float* Bn    = (float*)alloc((size_t)M_EDGES * 4);
  float* ef    = (float*)alloc((size_t)M_EDGES * DMODEL * 4);
  float* conv  = (float*)alloc((size_t)N_NODES * DMODEL * 4);
  float* x1    = (float*)alloc((size_t)N_NODES * DMODEL * 4);
  float* qkv   = (float*)alloc((size_t)N_NODES * 3 * DMODEL * 4);
  float* ctx   = (float*)alloc((size_t)N_NODES * DMODEL * 4);
  float* aout  = (float*)alloc((size_t)N_NODES * DMODEL * 4);
  float* x2    = (float*)alloc((size_t)N_NODES * DMODEL * 4);
  float* ffh   = (float*)alloc((size_t)N_NODES * DFF_ * 4);
  float* ffo   = (float*)alloc((size_t)N_NODES * DMODEL * 4);
  float* x3    = (float*)alloc((size_t)N_NODES * DMODEL * 4);
  // bf16 weight copies
  bf16_t* Wh_b  = (bf16_t*)alloc((size_t)DMODEL * DMODEL * 2);
  bf16_t* inw_b = (bf16_t*)alloc((size_t)3 * DMODEL * DMODEL * 2);
  bf16_t* outw_b= (bf16_t*)alloc((size_t)DMODEL * DMODEL * 2);
  bf16_t* ff1_bw= (bf16_t*)alloc((size_t)DFF_ * DMODEL * 2);
  bf16_t* ff2_bw= (bf16_t*)alloc((size_t)DMODEL * DFF_ * 2);

  // zero atomic accumulation targets every call (deterministic replay)
  hipMemsetAsync(den,  0, (size_t)N_NODES * 4, stream);
  hipMemsetAsync(Dn,   0, (size_t)N_NODES * 4, stream);
  hipMemsetAsync(Bn,   0, (size_t)M_EDGES * 4, stream);
  hipMemsetAsync(ef,   0, (size_t)M_EDGES * DMODEL * 4, stream);
  hipMemsetAsync(conv, 0, (size_t)N_NODES * DMODEL * 4, stream);
  fill_kernel<<<(N_NODES + 255) / 256, 256, 0, stream>>>(amax, -1e30f, N_NODES);

  // weight pre-conversion to bf16 (cheap; enables async LDS staging)
  {
    int n1 = DMODEL * DMODEL, n2 = 3 * DMODEL * DMODEL, n3 = DFF_ * DMODEL;
    cvt_bf16_kernel<<<(n1 + 255) / 256, 256, 0, stream>>>(Wh,    Wh_b,   n1);
    cvt_bf16_kernel<<<(n2 + 255) / 256, 256, 0, stream>>>(in_w,  inw_b,  n2);
    cvt_bf16_kernel<<<(n1 + 255) / 256, 256, 0, stream>>>(out_w, outw_b, n1);
    cvt_bf16_kernel<<<(n3 + 255) / 256, 256, 0, stream>>>(ff1_w, ff1_bw, n3);
    cvt_bf16_kernel<<<(n3 + 255) / 256, 256, 0, stream>>>(ff2_w, ff2_bw, n3);
  }

  // --- hypergraph conv ---
  gemm_bf16_kernel<false><<<dim3(N_NODES / 64, DMODEL / 128), 256, 0, stream>>>(
      x, Wh_b, nullptr, xl, N_NODES, DMODEL, DMODEL);
  gemm_bf16_kernel<false><<<dim3(M_EDGES / 64, DMODEL / 128), 256, 0, stream>>>(
      hattr, Wh_b, nullptr, el, M_EDGES, DMODEL, DMODEL);
  rowdot_kernel<<<N_NODES / 8, 256, 0, stream>>>(xl, att, sx, N_NODES);
  rowdot_kernel<<<M_EDGES / 8, 256, 0, stream>>>(el, att + DMODEL, se, M_EDGES);
  edge_score_kernel<<<NNZ_E / 256, 256, 0, stream>>>(node, edge, sx, se, ascr, Dn, Bn, amax);
  edge_exp_kernel<<<NNZ_E / 256, 256, 0, stream>>>(node, ascr, amax, aexp, den);
  scatter_edge_kernel<<<NNZ_E, 256, 0, stream>>>(node, edge, aexp, den, Bn, xl, ef);
  scatter_node_kernel<<<NNZ_E, 256, 0, stream>>>(node, edge, aexp, den, Dn, ef, conv);
  // x1 = LN(x + conv + bias_h)
  add_ln_kernel<<<N_NODES, 256, 0, stream>>>(x, conv, bias_h, ln1_g, ln1_b, x1);

  // --- transformer encoder layer (post-norm) ---
  gemm_bf16_kernel<false><<<dim3(N_NODES / 64, (3 * DMODEL) / 128), 256, 0, stream>>>(
      x1, inw_b, in_b, qkv, N_NODES, 3 * DMODEL, DMODEL);
  attn_kernel<<<dim3(N_NODES / 128, NHEAD), 256, 0, stream>>>(qkv, ctx);
  gemm_bf16_kernel<false><<<dim3(N_NODES / 64, DMODEL / 128), 256, 0, stream>>>(
      ctx, outw_b, out_b, aout, N_NODES, DMODEL, DMODEL);
  add_ln_kernel<<<N_NODES, 256, 0, stream>>>(x1, aout, nullptr, tln1_g, tln1_b, x2);
  gemm_bf16_kernel<true><<<dim3(N_NODES / 64, DFF_ / 128), 256, 0, stream>>>(
      x2, ff1_bw, ff1_b, ffh, N_NODES, DFF_, DMODEL);
  gemm_bf16_kernel<false><<<dim3(N_NODES / 64, DMODEL / 128), 256, 0, stream>>>(
      ffh, ff2_bw, ff2_b, ffo, N_NODES, DMODEL, DFF_);
  add_ln_kernel<<<N_NODES, 256, 0, stream>>>(x2, ffo, nullptr, tln2_g, tln2_b, x3);
  // out = LN(x1 + x3)
  add_ln_kernel<<<N_NODES, 256, 0, stream>>>(x1, x3, nullptr, ln2_g, ln2_b, out);
}
